// Generator_75651553952374
// MI455X (gfx1250) — matrix operand order repaired and checked by reading, not compiled, not run on hardware
//
#include <hip/hip_runtime.h>

// ---------- WMMA types (gfx1250, wave32) ----------
typedef _Float16 __attribute__((ext_vector_type(16))) v16h;
typedef float    __attribute__((ext_vector_type(8)))  v8f;

union AF {
  v16h v;
  _Float16 h[16];
  uint4 q[2];
};

// Load B fragment (32x16 f16), unconditional: B is zero-padded to Npad rows.
// lane lm = column n, half i -> k = k0 + lg*16 + i
__device__ __forceinline__ void load_bfrag(AF& f, const _Float16* __restrict__ Bm,
                                           int bn, int Kp, int k0, int lg) {
  const _Float16* p = Bm + (size_t)bn * Kp + k0 + lg * 16;
  f.q[0] = *(const uint4*)(p);
  f.q[1] = *(const uint4*)(p + 8);
}

// =====================================================================
// Dense f16 GEMM: out[M,N] = A[M,Kp] * B^T  (B: f16 [Npad][Kp], zero-pad).
// One wave computes a 16x32 tile (two 16x16 WMMA tiles sharing A).
// =====================================================================
__global__ __launch_bounds__(128) void k_gemm16(
    const _Float16* __restrict__ A, const _Float16* __restrict__ Bm,
    const float* __restrict__ bias, float biasScale, int doLeaky,
    float* __restrict__ o32, _Float16* __restrict__ o16,
    int M, int N, int Kp)
{
  const int wave = (blockIdx.x * blockDim.x + threadIdx.x) >> 5;
  const int lane = threadIdx.x & 31;
  const int nTT = (N + 31) >> 5;
  const int mT  = M >> 4;
  if (wave >= mT * nTT) return;  // uniform per wave
  const int tm = (wave / nTT) << 4;
  const int tn = (wave % nTT) << 5;
  const int lm = lane & 15;
  const int lg = lane >> 4;

  const _Float16* arow = A + (size_t)(tm + lm) * Kp;
  const int bn0 = tn + lm;
  const int bn1 = tn + 16 + lm;

  v8f acc0 = {}, acc1 = {};
  for (int k0 = 0; k0 < Kp; k0 += 32) {
    AF a, b0, b1;
    // A fragment 16x32: half i -> k = k0 + lg*16 + i
    const _Float16* p = arow + k0 + lg * 16;
    a.q[0] = *(const uint4*)(p);
    a.q[1] = *(const uint4*)(p + 8);
    load_bfrag(b0, Bm, bn0, Kp, k0, lg);
    load_bfrag(b1, Bm, bn1, Kp, k0, lg);
    acc0 = __builtin_amdgcn_wmma_f32_16x16x32_f16(false, a.v, false, b0.v,
                                                  (short)0, acc0, false, false);
    acc1 = __builtin_amdgcn_wmma_f32_16x16x32_f16(false, a.v, false, b1.v,
                                                  (short)0, acc1, false, false);
  }

#pragma unroll
  for (int half = 0; half < 2; ++half) {
    const int bn = half ? bn1 : bn0;
    if (bn >= N) continue;
#pragma unroll
    for (int v = 0; v < 8; ++v) {
      const int m = tm + lg * 8 + v;
      float r = half ? acc1[v] : acc0[v];
      if (bias) r += bias[bn] * biasScale;
      if (doLeaky) r = (r > 0.0f) ? r : 0.2f * r;
      const size_t oi = (size_t)m * N + bn;
      if (o32) o32[oi] = r;
      if (o16) o16[oi] = (_Float16)r;
    }
  }
}

// =====================================================================
// Implicit-im2col conv GEMM (3x3, pad 1). X: f16 NCHW (pre-modulated).
// K-order: k = kk*C + c (C power of two). B: f16 [Npad][Kp], zero-padded
// in both K and N (Npad multiple of 64). One wave = 16x64 tile:
// 4 WMMA tiles share one (expensive) im2col A fragment.
// =====================================================================
__global__ __launch_bounds__(128) void k_gemm_conv(
    const _Float16* __restrict__ X, const _Float16* __restrict__ Bm,
    const float* __restrict__ dsc, const float* __restrict__ bias,
    float biasScale, int doLeaky, _Float16* __restrict__ o16,
    int M, int N, int K, int Kp,
    int C, int cshift, int H, int W, int pshift, int wshift)
{
  const int wave = (blockIdx.x * blockDim.x + threadIdx.x) >> 5;
  const int lane = threadIdx.x & 31;
  const int nQT = (N + 63) >> 6;
  const int mT  = M >> 4;
  if (wave >= mT * nQT) return;  // uniform per wave
  const int tm = (wave / nQT) << 4;
  const int tn = (wave % nQT) << 6;
  const int lm = lane & 15;
  const int lg = lane >> 4;
  const int P  = 1 << pshift;

  // pixel decode for this lane's A row
  const int am = tm + lm;
  const int bb = am >> pshift;
  const int pp = am & (P - 1);
  const int ph = pp >> wshift;
  const int pw = pp & (W - 1);
  const _Float16* Xb = X + ((size_t)bb << (pshift + cshift));  // sample base

  v8f acc[4] = {{}, {}, {}, {}};
  for (int k0 = 0; k0 < Kp; k0 += 32) {
    AF a;
    // branchless im2col gather: clamped address + value select
#pragma unroll
    for (int i = 0; i < 16; ++i) {
      const int k  = k0 + lg * 16 + i;
      const int c  = k & (C - 1);
      const int kk = k >> cshift;
      const int kh = kk / 3;
      const int kw = kk - kh * 3;
      const int ih = ph + kh - 1;
      const int iw = pw + kw - 1;
      const int ihc = min(max(ih, 0), H - 1);
      const int iwc = min(max(iw, 0), W - 1);
      const _Float16 t =
          Xb[((size_t)c << pshift) + (ihc << wshift) + iwc];
      const bool ok = (k < K) & ((unsigned)ih < (unsigned)H) &
                      ((unsigned)iw < (unsigned)W);
      a.h[i] = ok ? t : (_Float16)0.0f;
    }
#pragma unroll
    for (int j = 0; j < 4; ++j) {
      AF b;
      load_bfrag(b, Bm, tn + j * 16 + lm, Kp, k0, lg);
      acc[j] = __builtin_amdgcn_wmma_f32_16x16x32_f16(
          false, a.v, false, b.v, (short)0, acc[j], false, false);
    }
  }

#pragma unroll
  for (int j = 0; j < 4; ++j) {
    const int bn = tn + j * 16 + lm;
    if (bn >= N) continue;
#pragma unroll
    for (int v = 0; v < 8; ++v) {
      const int m  = tm + lg * 8 + v;
      const int mb = m >> pshift;
      const int mp = m & (P - 1);
      float r = acc[j][v];
      if (dsc)  r *= dsc[(size_t)mb * N + bn];
      if (bias) r += bias[bn] * biasScale;
      if (doLeaky) r = (r > 0.0f) ? r : 0.2f * r;
      o16[((size_t)mb * N + bn) * P + mp] = (_Float16)r;
    }
  }
}

// ---------- weight converters (f32 -> f16, zero-padded rows; tiny) ----------
// [N][K] row-major -> f16 [Npad][K]
__global__ void k_cvt_lin(const float* __restrict__ in, _Float16* __restrict__ out,
                          int totalIn, int totalOut, float scale) {
  const int i = blockIdx.x * blockDim.x + threadIdx.x;
  if (i >= totalOut) return;
  out[i] = (i < totalIn) ? (_Float16)(in[i] * scale) : (_Float16)0.0f;
}
// init_ct [K][N] -> f16 [N][K] (N multiple of 32 already)
__global__ void k_cvt_tr(const float* __restrict__ in, _Float16* __restrict__ out,
                         int K, int N) {
  const int i = blockIdx.x * blockDim.x + threadIdx.x;
  if (i >= N * K) return;
  const int n = i / K, k = i - n * K;
  out[i] = (_Float16)in[(size_t)k * N + n];
}
// OIHW [N][C][3][3] -> f16 [Npad][Kp], k = kk*C + c, zero-padded K and N
__global__ void k_cvt_conv(const float* __restrict__ in, _Float16* __restrict__ out,
                           int N, int C, int Kp, int Npad) {
  const int i = blockIdx.x * blockDim.x + threadIdx.x;
  if (i >= Npad * Kp) return;
  const int n = i / Kp, k = i - n * Kp;
  _Float16 v = (_Float16)0.0f;
  if (n < N && k < 9 * C) {
    const int kk = k / C, c = k - kk * C;
    v = (_Float16)in[((size_t)n * C + c) * 9 + kk];
  }
  out[i] = v;
}

// ---------- pre-modulate activations: xm = x * (g[b,c] + 1) ----------
__global__ void k_mod(const _Float16* __restrict__ in, const float* __restrict__ g,
                      _Float16* __restrict__ out, int total, int pshift, int cshift) {
  const int i = blockIdx.x * blockDim.x + threadIdx.x;
  if (i >= total) return;
  const int c = (i >> pshift) & ((1 << cshift) - 1);
  const int b = i >> (pshift + cshift);
  out[i] = (_Float16)((float)in[i] * (g[((size_t)b << cshift) + c] + 1.0f));
}

// ---------- normalize x over dim1 (1024) and write f16 ----------
__global__ __launch_bounds__(256) void k_norm(const float* __restrict__ x,
                                              _Float16* __restrict__ o) {
  __shared__ float red[256];
  const int b = blockIdx.x, t = threadIdx.x;
  float s = 0.0f;
  for (int i = t; i < 1024; i += 256) { const float v = x[b * 1024 + i]; s += v * v; }
  red[t] = s;
  __syncthreads();
  for (int st = 128; st > 0; st >>= 1) {
    if (t < st) red[t] += red[t + st];
    __syncthreads();
  }
  const float inv = 1.0f / fmaxf(sqrtf(red[0]), 1e-12f);
  for (int i = t; i < 1024; i += 256)
    o[b * 1024 + i] = (_Float16)(x[b * 1024 + i] * inv);
}

// ---------- per-(o,c) sum of squared 3x3 weights ----------
__global__ void k_wsq(const float* __restrict__ Wc, float* __restrict__ S, int OC) {
  const int i = blockIdx.x * blockDim.x + threadIdx.x;
  if (i >= OC) return;
  const float* p = Wc + (size_t)i * 9;
  float s = 0.0f;
  for (int j = 0; j < 9; ++j) s += p[j] * p[j];
  S[i] = s;
}

// ---------- demod: d[b,o] = rsqrt( sum_c (g+1)^2 * S[o,c] + 1e-8 ) ----------
__global__ void k_demod(const float* __restrict__ g, const float* __restrict__ S,
                        float* __restrict__ d, int Bsz, int O, int C) {
  const int i = blockIdx.x * blockDim.x + threadIdx.x;
  if (i >= Bsz * O) return;
  const int b = i / O, o = i - b * O;
  float s = 0.0f;
  for (int c = 0; c < C; ++c) {
    const float t = g[(size_t)b * C + c] + 1.0f;
    s += t * t * S[(size_t)o * C + c];
  }
  d[i] = rsqrtf(s + 1e-8f);
}

// ---------- RGB 1x1 modulated conv (no demod) + accumulate ----------
__global__ void k_rgb(const _Float16* __restrict__ xi, const float* __restrict__ gr,
                      const float* __restrict__ rc, const float* __restrict__ prev,
                      float* __restrict__ out, int Bsz, int C, int P) {
  const int i = blockIdx.x * blockDim.x + threadIdx.x;
  if (i >= Bsz * P) return;
  const int b = i / P, p = i - b * P;
  float s = 0.0f;
  for (int c = 0; c < C; ++c)
    s += (float)xi[((size_t)b * C + c) * P + p] * (gr[(size_t)b * C + c] + 1.0f) * rc[c];
  out[i] = s + (prev ? prev[i] : 0.0f);
}

// ---------- 2x bilinear upsample, align_corners=False ----------
template <typename T>
__global__ void k_up2(const T* __restrict__ in, T* __restrict__ out,
                      int Bc, int H, int W) {
  const int H2 = 2 * H, W2 = 2 * W;
  const size_t total = (size_t)Bc * H2 * W2;
  const size_t idx = (size_t)blockIdx.x * blockDim.x + threadIdx.x;
  if (idx >= total) return;
  const int x2 = (int)(idx % W2);
  const int y2 = (int)((idx / W2) % H2);
  const int bc = (int)(idx / ((size_t)W2 * H2));
  const float fy = y2 * 0.5f - 0.25f, fx = x2 * 0.5f - 0.25f;
  const int y0 = (int)floorf(fy), x0 = (int)floorf(fx);
  const float wy = fy - y0, wx = fx - x0;
  const int y0c = min(max(y0, 0), H - 1), y1c = min(max(y0 + 1, 0), H - 1);
  const int x0c = min(max(x0, 0), W - 1), x1c = min(max(x0 + 1, 0), W - 1);
  const T* p = in + (size_t)bc * H * W;
  const float v00 = (float)p[y0c * W + x0c], v01 = (float)p[y0c * W + x1c];
  const float v10 = (float)p[y1c * W + x0c], v11 = (float)p[y1c * W + x1c];
  out[idx] = (T)((v00 * (1.0f - wx) + v01 * wx) * (1.0f - wy) +
                 (v10 * (1.0f - wx) + v11 * wx) * wy);
}

// ---------- tanh ----------
__global__ void k_tanh(const float* __restrict__ in, float* __restrict__ out, int n) {
  const int i = blockIdx.x * blockDim.x + threadIdx.x;
  if (i < n) out[i] = tanhf(in[i]);
}

// =====================================================================
// Host side
// =====================================================================
static inline int lg2i(int v) { int s = 0; while ((1 << s) < v) ++s; return s; }

static inline void gemm16(hipStream_t s, const _Float16* A, const _Float16* Bm,
                          const float* bias, float bScale, int leaky,
                          float* o32, _Float16* o16, int M, int N, int Kp) {
  const int waves = (M / 16) * ((N + 31) / 32);
  k_gemm16<<<(waves + 3) / 4, 128, 0, s>>>(A, Bm, bias, bScale, leaky, o32, o16,
                                           M, N, Kp);
}

static inline void gemmConv(hipStream_t s, const _Float16* X, const _Float16* Bm,
                            const float* dsc, const float* bias, float bScale,
                            int leaky, _Float16* o16, int N, int C, int H, int W) {
  const int P = H * W, M = 256 * P;
  const int K = 9 * C, Kp = (K + 31) & ~31;
  const int waves = (M / 16) * ((N + 63) / 64);
  k_gemm_conv<<<(waves + 3) / 4, 128, 0, s>>>(X, Bm, dsc, bias, bScale, leaky, o16,
                                              M, N, K, Kp, C, lg2i(C), H, W,
                                              lg2i(P), lg2i(W));
}

extern "C" void kernel_launch(void* const* d_in, const int* in_sizes, int n_in,
                              void* d_out, int out_size, void* d_ws, size_t ws_size,
                              hipStream_t stream) {
  (void)in_sizes; (void)n_in; (void)out_size; (void)ws_size;
  const int B = 256, L = 128;

  // ---- inputs in setup_inputs() dict (insertion) order ----
  int ix = 0;
  const float* x = (const float*)d_in[ix++];
  const float *mapW[4], *mapB[4];
  for (int i = 0; i < 4; ++i) { mapW[i] = (const float*)d_in[ix++]; mapB[i] = (const float*)d_in[ix++]; }
  const float* initCT = (const float*)d_in[ix++];
  const float* icW = (const float*)d_in[ix++];
  const float* icB = (const float*)d_in[ix++];
  struct Blk { const float *s1W, *s1b, *c1, *s2W, *s2b, *c2, *rsW, *rsb, *rc; } blk[4];
  for (int i = 0; i < 4; ++i) {
    blk[i].s1W = (const float*)d_in[ix++]; blk[i].s1b = (const float*)d_in[ix++];
    blk[i].c1  = (const float*)d_in[ix++];
    blk[i].s2W = (const float*)d_in[ix++]; blk[i].s2b = (const float*)d_in[ix++];
    blk[i].c2  = (const float*)d_in[ix++];
    blk[i].rsW = (const float*)d_in[ix++]; blk[i].rsb = (const float*)d_in[ix++];
    blk[i].rc  = (const float*)d_in[ix++];
  }

  // ---- workspace carve ----
  char* wp = (char*)d_ws;
  auto alloc = [&](size_t bytes) -> void* {
    void* p = (void*)wp;
    wp += (bytes + 255) & ~(size_t)255;
    return p;
  };
  _Float16* xn   = (_Float16*)alloc((size_t)B * 1024 * 2);
  _Float16* h0   = (_Float16*)alloc((size_t)B * L * 2);
  _Float16* h1   = (_Float16*)alloc((size_t)B * L * 2);
  _Float16* wlat = (_Float16*)alloc((size_t)B * L * 2);
  float* g1   = (float*)alloc((size_t)B * 128 * 4);
  float* g2   = (float*)alloc((size_t)B * 128 * 4);
  float* gr   = (float*)alloc((size_t)B * 128 * 4);
  float* dd1  = (float*)alloc((size_t)B * 128 * 4);
  float* dd2  = (float*)alloc((size_t)B * 128 * 4);
  float* Ssum = (float*)alloc((size_t)128 * 128 * 4);
  float* rgb0 = (float*)alloc((size_t)B * 1024 * 4);
  float* rgb1 = (float*)alloc((size_t)B * 1024 * 4);
  _Float16* wk16 = (_Float16*)alloc((size_t)2048 * 128 * 2);  // weight scratch
  _Float16* bufs[3];
  for (int i = 0; i < 3; ++i) bufs[i] = (_Float16*)alloc((size_t)33554432);

  // convert linear weight [N][K] -> padded [Npad][K]
  auto cvtLin = [&](const float* w, int N, int K, float sc) {
    const int Npad = (N + 31) & ~31;
    const int tot = Npad * K;
    k_cvt_lin<<<(tot + 255) / 256, 256, 0, stream>>>(w, wk16, N * K, tot, sc);
  };
  // convert conv weight OIHW -> padded reordered [Npad][Kp]
  auto cvtConv = [&](const float* w, int N, int C) {
    const int Kp = (9 * C + 31) & ~31;
    const int Npad = (N + 63) & ~63;
    const int tot = Npad * Kp;
    k_cvt_conv<<<(tot + 255) / 256, 256, 0, stream>>>(w, wk16, N, C, Kp, Npad);
  };

  // ---- 1. mapping network ----
  k_norm<<<B, 256, 0, stream>>>(x, xn);
  cvtLin(mapW[0], 128, 1024, 0.1f);
  gemm16(stream, xn, wk16, mapB[0], 0.1f, 1, nullptr, h0, B, L, 1024);
  cvtLin(mapW[1], 128, 128, 0.1f);
  gemm16(stream, h0, wk16, mapB[1], 0.1f, 1, nullptr, h1, B, L, 128);
  cvtLin(mapW[2], 128, 128, 0.1f);
  gemm16(stream, h1, wk16, mapB[2], 0.1f, 1, nullptr, h0, B, L, 128);
  cvtLin(mapW[3], 128, 128, 0.1f);
  gemm16(stream, h0, wk16, mapB[3], 0.1f, 1, nullptr, wlat, B, L, 128);

  // ---- 2. initial einsum: w[b,i] * ct[i,o,h,w] -> xi [B,128,4,4] ----
  k_cvt_tr<<<(2048 * 128 + 255) / 256, 256, 0, stream>>>(initCT, wk16, 128, 2048);
  gemm16(stream, wlat, wk16, nullptr, 0.0f, 0, nullptr, bufs[0], B, 2048, 128);

  // ---- 3. initial 3x3 conv (128->128, +bias, no act) ----
  cvtConv(icW, 128, 128);
  gemmConv(stream, bufs[0], wk16, nullptr, icB, 1.0f, 0, bufs[1], 128, 128, 4, 4);
  int cur = 1;

  // ---- 4. generator blocks ----
  const int cfg_ic[4] = {128, 128, 128, 64};
  const int cfg_oc[4] = {128, 128, 64, 1};
  const int cfg_up[4] = {0, 1, 1, 1};
  const int cfg_ur[4] = {1, 1, 1, 0};
  float* rgbCur = rgb0;
  float* rgbOth = rgb1;
  const float* rgbPrev = nullptr;
  int H = 4, W = 4;

  for (int bi = 0; bi < 4; ++bi) {
    const int ic = cfg_ic[bi], oc = cfg_oc[bi];

    // styles (no activation)
    cvtLin(blk[bi].s1W, ic, 128, 1.0f);
    gemm16(stream, wlat, wk16, blk[bi].s1b, 1.0f, 0, g1, nullptr, B, ic, 128);
    cvtLin(blk[bi].s2W, oc, 128, 1.0f);
    gemm16(stream, wlat, wk16, blk[bi].s2b, 1.0f, 0, g2, nullptr, B, oc, 128);
    cvtLin(blk[bi].rsW, oc, 128, 1.0f);
    gemm16(stream, wlat, wk16, blk[bi].rsb, 1.0f, 0, gr, nullptr, B, oc, 128);

    if (cfg_up[bi]) {
      const int nx = (cur + 1) % 3;
      const size_t tot = (size_t)B * ic * (2 * H) * (2 * W);
      k_up2<_Float16><<<(int)((tot + 255) / 256), 256, 0, stream>>>(
          bufs[cur], bufs[nx], B * ic, H, W);
      cur = nx;
      H *= 2; W *= 2;
    }
    const int P = H * W;

    // conv1: modulate -> implicit GEMM (demod, leaky)
    {
      const int t0 = (cur + 1) % 3, t1 = (cur + 2) % 3;
      const int tot = B * ic * P;
      k_mod<<<(tot + 255) / 256, 256, 0, stream>>>(bufs[cur], g1, bufs[t0], tot,
                                                   lg2i(P), lg2i(ic));
      k_wsq<<<(oc * ic + 255) / 256, 256, 0, stream>>>(blk[bi].c1, Ssum, oc * ic);
      k_demod<<<(B * oc + 255) / 256, 256, 0, stream>>>(g1, Ssum, dd1, B, oc, ic);
      cvtConv(blk[bi].c1, oc, ic);
      gemmConv(stream, bufs[t0], wk16, dd1, nullptr, 0.0f, 1, bufs[t1], oc, ic, H, W);
      cur = t1;
    }
    // conv2: modulate -> implicit GEMM (demod, leaky)
    {
      const int t0 = (cur + 1) % 3, t1 = (cur + 2) % 3;
      const int tot = B * oc * P;
      k_mod<<<(tot + 255) / 256, 256, 0, stream>>>(bufs[cur], g2, bufs[t0], tot,
                                                   lg2i(P), lg2i(oc));
      k_wsq<<<(oc * oc + 255) / 256, 256, 0, stream>>>(blk[bi].c2, Ssum, oc * oc);
      k_demod<<<(B * oc + 255) / 256, 256, 0, stream>>>(g2, Ssum, dd2, B, oc, oc);
      cvtConv(blk[bi].c2, oc, oc);
      gemmConv(stream, bufs[t0], wk16, dd2, nullptr, 0.0f, 1, bufs[t1], oc, oc, H, W);
      cur = t1;
    }

    // RGB 1x1 mod conv (no demod) + accumulate
    k_rgb<<<(B * P + 255) / 256, 256, 0, stream>>>(bufs[cur], gr, blk[bi].rc,
                                                   rgbPrev, rgbCur, B, oc, P);
    rgbPrev = rgbCur;
    if (cfg_ur[bi]) {
      const size_t tot = (size_t)B * (2 * H) * (2 * W);
      k_up2<float><<<(int)((tot + 255) / 256), 256, 0, stream>>>(rgbCur, rgbOth, B, H, W);
      float* t = rgbCur; rgbCur = rgbOth; rgbOth = t;
      rgbPrev = rgbCur;
    }
  }

  // ---- 5. tanh -> output [256,1,32,32] f32 ----
  const int n = B * 32 * 32;
  k_tanh<<<(n + 255) / 256, 256, 0, stream>>>(rgbCur, (float*)d_out, n);
}
